// sLSTMCell_21062519619836
// MI455X (gfx1250) — matrix operand order repaired and checked by reading, not compiled, not run on hardware
//
#include <hip/hip_runtime.h>
#include <hip/hip_bf16.h>

typedef __attribute__((ext_vector_type(16))) __bf16 bf16x16;
typedef __attribute__((ext_vector_type(8)))  float  floatx8;

union Frag {
    uint4   u[2];
    bf16x16 v;
};

__device__ inline unsigned short f32_to_bf16(float f) {
    union { float f; unsigned int u; } a; a.f = f;
    unsigned int u = a.u;
    unsigned int r = u + 0x7FFFu + ((u >> 16) & 1u);   // round-to-nearest-even
    return (unsigned short)(r >> 16);
}

// Async global->LDS copy, 16 B per lane, tracked by ASYNCcnt (CDNA5 path).
#define ASYNC_B128(ldsaddr, gaddr, IMM)                                        \
    asm volatile("global_load_async_to_lds_b128 %0, %1, off offset:" #IMM      \
                 :: "v"(ldsaddr), "v"(gaddr) : "memory")

#define WAIT_ASYNC0() asm volatile("s_wait_asynccnt 0x0" ::: "memory")

// -------------------------------------------------------------------------
// Phase 1a: A = [x | h_prev] converted to bf16, row-major [8192][4096]
// -------------------------------------------------------------------------
__global__ __launch_bounds__(256)
void convert_a(const float* __restrict__ x, const float* __restrict__ h,
               unsigned short* __restrict__ A) {
    int tid = blockIdx.x * 256 + threadIdx.x;     // each thread: 4 elements
    int e   = tid * 4;
    int b   = e >> 12;                            // / 4096
    int k   = e & 4095;
    const float* src = (k < 2048) ? (x + (size_t)b * 2048 + k)
                                  : (h + (size_t)b * 2048 + (k - 2048));
    float4 v = *(const float4*)src;
    unsigned int lo = (unsigned int)f32_to_bf16(v.x) | ((unsigned int)f32_to_bf16(v.y) << 16);
    unsigned int hi = (unsigned int)f32_to_bf16(v.z) | ((unsigned int)f32_to_bf16(v.w) << 16);
    uint2 o; o.x = lo; o.y = hi;
    *(uint2*)(A + (size_t)e) = o;
}

// -------------------------------------------------------------------------
// Phase 1b: Bt[n][k] bf16, n = gate*2048 + j, k in [0,4096)
//           k <  2048 : W_gate[k][j] ; k >= 2048 : R_gate[k-2048][j]
// Tiled 32x32 transpose through LDS (coalesced read + coalesced write).
// -------------------------------------------------------------------------
__global__ __launch_bounds__(256)
void transpose_w(const float* __restrict__ w0, const float* __restrict__ w1,
                 const float* __restrict__ w2, const float* __restrict__ w3,
                 const float* __restrict__ r0, const float* __restrict__ r1,
                 const float* __restrict__ r2, const float* __restrict__ r3,
                 unsigned short* __restrict__ Bt) {
    __shared__ unsigned short tile[32][33];
    int tn = blockIdx.x & 255;          // 256 n-tiles of 32
    int tk = blockIdx.x >> 8;           // 128 k-tiles of 32
    int n0 = tn * 32, k0 = tk * 32;
    int g  = n0 >> 11;                  // / 2048
    int j0 = n0 & 2047;
    const float* wsrc = (g == 0) ? w0 : (g == 1) ? w1 : (g == 2) ? w2 : w3;
    const float* rsrc = (g == 0) ? r0 : (g == 1) ? r1 : (g == 2) ? r2 : r3;
    const float* src; int kbase;
    if (k0 < 2048) { src = wsrc; kbase = k0; } else { src = rsrc; kbase = k0 - 2048; }

    int tx = threadIdx.x & 31, ty = threadIdx.x >> 5;   // ty 0..7
    for (int s = 0; s < 4; ++s) {
        int kk = ty + 8 * s;
        tile[kk][tx] = f32_to_bf16(src[(size_t)(kbase + kk) * 2048 + j0 + tx]);
    }
    __syncthreads();
    for (int s = 0; s < 4; ++s) {
        int nn = ty + 8 * s;
        Bt[(size_t)(n0 + nn) * 4096 + k0 + tx] = tile[tx][nn];
    }
}

// -------------------------------------------------------------------------
// Init: out[b] = fc_b  (GEMM epilogue accumulates h_t @ fc_w atomically)
// -------------------------------------------------------------------------
__global__ void init_out(float* __restrict__ out0, const float* __restrict__ fc_b) {
    int i = blockIdx.x * 256 + threadIdx.x;
    if (i < 8192) out0[i] = fc_b[0];
}

// -------------------------------------------------------------------------
// Phase 2: fused 4-gate GEMM + sLSTM pointwise + fc reduction
//   block: 256 threads = 8 waves (4 row-waves x 2 col-waves)
//   tile : BM=128 rows, 64 cols per gate (x4 gates), BK=32
//   double-buffered LDS filled by async global->LDS (ASYNCcnt) copies
//   K-loop kept at unroll 1 so the live set stays < 256 VGPRs (no MSB mode)
// -------------------------------------------------------------------------
#define BM    128
#define BNG   64
#define BK    32
#define PITCH 40          // bf16 pitch (32 + 8 pad), row stride 80 B (16-aligned)

__global__ __launch_bounds__(256)
void slstm_gemm(const unsigned short* __restrict__ A,    // [8192][4096] bf16
                const unsigned short* __restrict__ Bt,   // [8192][4096] bf16
                const float* __restrict__ bi,  const float* __restrict__ bfv,
                const float* __restrict__ bo,  const float* __restrict__ bz,
                const float* __restrict__ c_prev, const float* __restrict__ n_prev,
                const float* __restrict__ m_prev, const float* __restrict__ fc_w,
                float* __restrict__ out0, float* __restrict__ outH,
                float* __restrict__ outC, float* __restrict__ outN,
                float* __restrict__ outM) {
    constexpr int KTOT = 4096;
    __shared__ __align__(16) unsigned short sA[2][BM * PITCH];
    __shared__ __align__(16) unsigned short sB[2][4 * BNG * PITCH];

    const int bcols = 2048 / BNG;                 // 32
    int brow = blockIdx.x / bcols;
    int bcol = blockIdx.x % bcols;
    int row0 = brow * BM;
    int col0 = bcol * BNG;

    int t    = threadIdx.x;
    int lane = t & 31;
    int w    = t >> 5;
    int wr   = w >> 1, wc = w & 1;                // 4 x 2 wave grid
    int lm   = lane & 15, lh = lane >> 4;

    // ---- async global->LDS staging addresses (per thread) ----
    int arow = t >> 1;                            // 0..127
    int akp  = (t & 1) * 16;                      // bf16 offset within K-slab
    const unsigned short* ga = A + (size_t)(row0 + arow) * KTOT + akp;
    int gidx = t >> 6, jn = t & 63;               // one Bt row per thread
    const unsigned short* gb = Bt + (size_t)(gidx * 2048 + col0 + jn) * KTOT;

    unsigned ldsA[2], ldsB[2];
    ldsA[0] = (unsigned)(size_t)&sA[0][arow * PITCH + akp];
    ldsA[1] = (unsigned)(size_t)&sA[1][arow * PITCH + akp];
    ldsB[0] = (unsigned)(size_t)&sB[0][t * PITCH];
    ldsB[1] = (unsigned)(size_t)&sB[1][t * PITCH];

    floatx8 acc[4][2][2] = {};                    // [gate][mi][ni]

    // prefetch K-step 0 into buffer 0 (async, no VGPR round-trip)
    ASYNC_B128(ldsA[0], ga, 0);
    ASYNC_B128(ldsA[0], ga, 16);
    ASYNC_B128(ldsB[0], gb, 0);
    ASYNC_B128(ldsB[0], gb, 16);
    ASYNC_B128(ldsB[0], gb, 32);
    ASYNC_B128(ldsB[0], gb, 48);

    const int NK = KTOT / BK;                     // 128
#pragma unroll 1
    for (int kt = 0; kt < NK; ++kt) {
        int cur = kt & 1, nxt = cur ^ 1;

        WAIT_ASYNC0();                            // my writes into 'cur' done
        __syncthreads();                          // everyone's writes visible

        if (kt + 1 < NK) {                        // fill 'nxt' asynchronously
            const unsigned short* ga2 = ga + (size_t)(kt + 1) * BK;
            const unsigned short* gb2 = gb + (size_t)(kt + 1) * BK;
            ASYNC_B128(ldsA[nxt], ga2, 0);
            ASYNC_B128(ldsA[nxt], ga2, 16);
            ASYNC_B128(ldsB[nxt], gb2, 0);
            ASYNC_B128(ldsB[nxt], gb2, 16);
            ASYNC_B128(ldsB[nxt], gb2, 32);
            ASYNC_B128(ldsB[nxt], gb2, 48);
        }

        const unsigned short* sAc = &sA[cur][0];
        const unsigned short* sBc = &sB[cur][0];

        // A fragments: lane l -> M = l%16, elems 0-7: K=(l>>4)*8.., 8-15: +16
        Frag af[2];
#pragma unroll
        for (int mi = 0; mi < 2; ++mi) {
            int m = wr * 32 + mi * 16 + lm;
            const unsigned short* p = sAc + m * PITCH + lh * 8;
            af[mi].u[0] = *(const uint4*)p;
            af[mi].u[1] = *(const uint4*)(p + 16);
        }
#pragma unroll
        for (int g = 0; g < 4; ++g) {
            // B fragments: lane l -> N = l%16, elems 0-7: K=(l>>4)*8.., 8-15: +16
            Frag bfr[2];
#pragma unroll
            for (int ni = 0; ni < 2; ++ni) {
                int n = g * BNG + wc * 32 + ni * 16 + lm;
                const unsigned short* p = sBc + n * PITCH + lh * 8;
                bfr[ni].u[0] = *(const uint4*)p;
                bfr[ni].u[1] = *(const uint4*)(p + 16);
            }
#pragma unroll
            for (int mi = 0; mi < 2; ++mi)
#pragma unroll
                for (int ni = 0; ni < 2; ++ni)
                    acc[g][mi][ni] = __builtin_amdgcn_wmma_f32_16x16x32_bf16(
                        false, af[mi].v, false, bfr[ni].v,
                        (short)0, acc[g][mi][ni], false, false);
        }
    }

    // ---------------- fused sLSTM epilogue ----------------
    int jbase = col0 + wc * 32;
    for (int mi = 0; mi < 2; ++mi) {
        for (int v = 0; v < 8; ++v) {
            int brow_ = row0 + wr * 32 + mi * 16 + v + 8 * lh;   // C layout: M = v + 8*(l>>4)
            float partial = 0.f;
            for (int ni = 0; ni < 2; ++ni) {
                int j = jbase + ni * 16 + lm;                    // N = l%16
                size_t idx = (size_t)brow_ * 2048 + j;
                float it = acc[0][mi][ni][v] + bi[j];
                float ft = acc[1][mi][ni][v] + bfv[j];
                float ot = acc[2][mi][ni][v] + bo[j];
                float zt = acc[3][mi][ni][v] + bz[j];

                float fsig  = 1.f / (1.f + __expf(-ft));
                float osig  = 1.f / (1.f + __expf(-ot));
                float ztan  = tanhf(zt);
                float logf_ = __logf(fsig);
                float mp    = m_prev[idx];
                float mt    = fmaxf(logf_ + mp, it);
                float ip    = __expf(it - mt);
                float fp    = __expf(logf_ + mp - mt);
                float ct    = fp * c_prev[idx] + ip * ztan;
                float nt    = fp * n_prev[idx] + ip;
                float ht    = osig * ct / nt;

                outH[idx] = ht;  outC[idx] = ct;
                outN[idx] = nt;  outM[idx] = mt;
                partial += ht * fc_w[j];
            }
            // rows live in lanes sharing (l>>4); reduce across the 16 N-lanes
            for (int off = 8; off; off >>= 1)
                partial += __shfl_xor(partial, off, 16);
            if (lm == 0) atomicAdd(&out0[brow_], partial);
        }
    }
}

// -------------------------------------------------------------------------
extern "C" void kernel_launch(void* const* d_in, const int* in_sizes, int n_in,
                              void* d_out, int out_size, void* d_ws, size_t ws_size,
                              hipStream_t stream) {
    const float* x      = (const float*)d_in[0];
    const float* h_prev = (const float*)d_in[1];
    const float* c_prev = (const float*)d_in[2];
    const float* n_prev = (const float*)d_in[3];
    const float* m_prev = (const float*)d_in[4];
    const float* w_i    = (const float*)d_in[5];
    const float* w_f    = (const float*)d_in[6];
    const float* w_o    = (const float*)d_in[7];
    const float* w_z    = (const float*)d_in[8];
    const float* r_i    = (const float*)d_in[9];
    const float* r_f    = (const float*)d_in[10];
    const float* r_o    = (const float*)d_in[11];
    const float* r_z    = (const float*)d_in[12];
    const float* b_i    = (const float*)d_in[13];
    const float* b_f    = (const float*)d_in[14];
    const float* b_o    = (const float*)d_in[15];
    const float* b_z    = (const float*)d_in[16];
    const float* fc_w   = (const float*)d_in[17];
    const float* fc_b   = (const float*)d_in[18];

    unsigned short* A  = (unsigned short*)d_ws;                 // 64 MB bf16
    unsigned short* Bt = A + (size_t)8192 * 4096;               // 64 MB bf16

    float* out0 = (float*)d_out;                                // [8192]
    float* outH = out0 + 8192;                                  // [8192][2048]
    float* outC = outH + (size_t)8192 * 2048;
    float* outN = outC + (size_t)8192 * 2048;
    float* outM = outN + (size_t)8192 * 2048;

    convert_a  <<<(8192 * 4096 / 4) / 256, 256, 0, stream>>>(x, h_prev, A);
    transpose_w<<<128 * 256, 256, 0, stream>>>(w_i, w_f, w_o, w_z,
                                               r_i, r_f, r_o, r_z, Bt);
    init_out   <<<32, 256, 0, stream>>>(out0, fc_b);
    slstm_gemm <<<64 * 32, 256, 0, stream>>>(A, Bt, b_i, b_f, b_o, b_z,
                                             c_prev, n_prev, m_prev, fc_w,
                                             out0, outH, outC, outN, outM);
}
